// Self_Attention_84181359001844
// MI455X (gfx1250) — matrix operand order repaired and checked by reading, not compiled, not run on hardware
//
#include <hip/hip_runtime.h>

typedef __attribute__((ext_vector_type(16))) _Float16 v16h;
typedef __attribute__((ext_vector_type(8)))  _Float16 v8h;
typedef __attribute__((ext_vector_type(8)))  float    v8f;

namespace {
constexpr int Bn   = 4;
constexpr int Nn   = 4096;
constexpr int Din  = 128;
constexpr int E    = 64;      // d_out
constexpr int ROWS = Bn * Nn; // 16384
constexpr int WTS  = 136;     // padded d-stride of W^T in LDS
constexpr int VTS  = 72;      // padded key-stride of V^T block in LDS
constexpr int PWS  = 72;      // padded key-stride of P^T in LDS
constexpr int BK   = 64;      // keys per block iteration
// 1/sqrt(64) * log2(e): softmax runs in base 2 (native v_exp_f32)
constexpr float QSCALE = 0.125f * 1.4426950408889634f;
}

// A operand (16x32 f16): per lane 16 contiguous K-elements:
// K = half*16 + [0,16), matching the B operand order.
__device__ __forceinline__ v16h load_a(const _Float16* row, int half) {
  v8h lo = *(const v8h*)(row + half * 16);
  v8h hi = *(const v8h*)(row + half * 16 + 8);
  return __builtin_shufflevector(lo, hi, 0, 1, 2, 3, 4, 5, 6, 7,
                                 8, 9, 10, 11, 12, 13, 14, 15);
}

// B operand (32x16 f16): per lane 16 contiguous K-elements at fixed column.
__device__ __forceinline__ v16h load_b(const _Float16* p) {
  v8h lo = *(const v8h*)(p);
  v8h hi = *(const v8h*)(p + 8);
  return __builtin_shufflevector(lo, hi, 0, 1, 2, 3, 4, 5, 6, 7,
                                 8, 9, 10, 11, 12, 13, 14, 15);
}

// 16 consecutive f32 -> v16h (coalesced float4 loads + convert).
__device__ __forceinline__ v16h cvt16(const float* p) {
  float4 f0 = *(const float4*)(p);
  float4 f1 = *(const float4*)(p + 4);
  float4 f2 = *(const float4*)(p + 8);
  float4 f3 = *(const float4*)(p + 12);
  v16h r;
  r[0]  = (_Float16)f0.x; r[1]  = (_Float16)f0.y; r[2]  = (_Float16)f0.z; r[3]  = (_Float16)f0.w;
  r[4]  = (_Float16)f1.x; r[5]  = (_Float16)f1.y; r[6]  = (_Float16)f1.z; r[7]  = (_Float16)f1.w;
  r[8]  = (_Float16)f2.x; r[9]  = (_Float16)f2.y; r[10] = (_Float16)f2.z; r[11] = (_Float16)f2.w;
  r[12] = (_Float16)f3.x; r[13] = (_Float16)f3.y; r[14] = (_Float16)f3.z; r[15] = (_Float16)f3.w;
  return r;
}

__device__ __forceinline__ v8h cvt8(const float* p) {
  float4 f0 = *(const float4*)(p);
  float4 f1 = *(const float4*)(p + 4);
  v8h r;
  r[0] = (_Float16)f0.x; r[1] = (_Float16)f0.y; r[2] = (_Float16)f0.z; r[3] = (_Float16)f0.w;
  r[4] = (_Float16)f1.x; r[5] = (_Float16)f1.y; r[6] = (_Float16)f1.z; r[7] = (_Float16)f1.w;
  return r;
}

// Async cache->LDS copy, 16B per lane, tracked by ASYNCcnt.
// VGLOBAL operand order: vdst(LDS addr), vaddr(i32 offset), saddr(64-bit base).
// LDS byte address = low 32 bits of the generic pointer (ISA aperture rule).
__device__ __forceinline__ void async_ld_b128(const void* lds, int goff,
                                              const void* gbase) {
  asm volatile("global_load_async_to_lds_b128 %0, %1, %2"
               :
               : "v"((unsigned)(uintptr_t)lds), "v"(goff), "s"(gbase)
               : "memory");
}
__device__ __forceinline__ void wait_async0() {
  asm volatile("s_wait_asynccnt 0x0" ::: "memory");
}

// -------------------------------------------------------------------------
// Kernel 1: QKV projection.
//  Q,K computed as D=(XW)^T: packed v8h stores into natural [row][e].
//  V computed as D=XW: packed v8h stores into transposed V^T[b][e][row],
//  which is exactly what the attention kernel's LDS fill wants.
// -------------------------------------------------------------------------
__global__ __launch_bounds__(256) void qkv_kernel(
    const float* __restrict__ x, const float* __restrict__ w,
    _Float16* __restrict__ qh, _Float16* __restrict__ kh,
    _Float16* __restrict__ vtg)
{
  __shared__ __align__(16) _Float16 wt[3 * E * WTS]; // W^T [wi][e][d], 51 KB

  const int tid  = threadIdx.x;
  const int wave = tid >> 5;
  const int lane = tid & 31;
  const int half = lane >> 4;
  const int m    = lane & 15;
  const int kg   = half;
  const int n    = m;

  // Stage weights transposed: w[wi][d][e] -> wt[wi][e][d] as f16.
  for (int i = tid * 4; i < 3 * Din * E; i += 256 * 4) {
    float4 f = *(const float4*)(w + i);
    int wi  = i >> 13;
    int rem = i & 8191;
    int d   = rem >> 6;
    int e   = rem & 63;
    wt[(wi * E + e + 0) * WTS + d] = (_Float16)f.x;
    wt[(wi * E + e + 1) * WTS + d] = (_Float16)f.y;
    wt[(wi * E + e + 2) * WTS + d] = (_Float16)f.z;
    wt[(wi * E + e + 3) * WTS + d] = (_Float16)f.w;
  }

  const int rowbase = (blockIdx.x * 8 + wave) * 16;
  const float* xrow = x + (size_t)(rowbase + n) * Din; // n == m

  // X operands straight from global (no LDS): B = X^T and A = X.
  v16h bx[4], ax[4];
  #pragma unroll
  for (int ks = 0; ks < 4; ++ks) {
    bx[ks] = cvt16(xrow + ks * 32 + kg * 16);
    v8h lo = cvt8(xrow + ks * 32 + half * 16);
    v8h hi = cvt8(xrow + ks * 32 + half * 16 + 8);
    ax[ks] = __builtin_shufflevector(lo, hi, 0, 1, 2, 3, 4, 5, 6, 7,
                                     8, 9, 10, 11, 12, 13, 14, 15);
  }
  __syncthreads();

  // Q, K: transposed orientation, packed stores into natural [row][e].
  _Float16* qk[2] = {qh, kh};
  #pragma unroll
  for (int wi = 0; wi < 2; ++wi) {
    #pragma unroll
    for (int et = 0; et < 4; ++et) {
      v8f acc = {};
      #pragma unroll
      for (int ks = 0; ks < 4; ++ks) {
        v16h aw = load_a(wt + (wi * E + et * 16 + m) * WTS + ks * 32, half);
        acc = __builtin_amdgcn_wmma_f32_16x16x32_f16(
                  false, aw, false, bx[ks], (short)0, acc, false, false);
      }
      const float scale = (wi == 0) ? QSCALE : 1.0f;
      v8h hh;
      #pragma unroll
      for (int r = 0; r < 8; ++r) hh[r] = (_Float16)(acc[r] * scale);
      *(v8h*)(qk[wi] + (size_t)(rowbase + n) * E + et * 16 + 8 * half) = hh;
    }
  }

  // V: natural orientation, packed stores into transposed V^T[b][e][row].
  const int bb   = rowbase / Nn;
  const int rloc = rowbase - bb * Nn;
  _Float16* vb = vtg + (size_t)bb * E * Nn;
  #pragma unroll
  for (int et = 0; et < 4; ++et) {
    v8f acc = {};
    #pragma unroll
    for (int ks = 0; ks < 4; ++ks) {
      v16h bwv = load_b(wt + (2 * E + et * 16 + n) * WTS + ks * 32 + kg * 16);
      acc = __builtin_amdgcn_wmma_f32_16x16x32_f16(
                false, ax[ks], false, bwv, (short)0, acc, false, false);
    }
    v8h hh;
    #pragma unroll
    for (int r = 0; r < 8; ++r) hh[r] = (_Float16)acc[r];
    *(v8h*)(vb + (size_t)(et * 16 + n) * Nn + rloc + 8 * half) = hh;
  }
}

// -------------------------------------------------------------------------
// Kernel 2: flash attention, transposed score tiles (S^T = K Q^T), per-lane
// scalar softmax stats, double-buffered K/V^T blocks filled by ASYNC
// cache->LDS copies (no VGPR round trip), one barrier per iteration.
// 4 waves/block, 256 blocks.
// -------------------------------------------------------------------------
__global__ __launch_bounds__(128) void attn_kernel(
    const _Float16* __restrict__ qh, const _Float16* __restrict__ kh,
    const _Float16* __restrict__ vtg, float* __restrict__ out)
{
  __shared__ __align__(16) _Float16 kbuf[2][BK * E];      // K [key][e], 2x8 KB
  __shared__ __align__(16) _Float16 vbuf[2][E * VTS];     // V^T [e][key], 2x9 KB
  __shared__ __align__(16) _Float16 ps[4][16 * PWS];      // per-wave P^T, 9 KB

  const int tid  = threadIdx.x;
  const int wave = tid >> 5;
  const int lane = tid & 31;
  const int half = lane >> 4;
  const int m    = lane & 15;
  const int kg   = half;
  const int n    = m;

  const int rowbase = (blockIdx.x * 4 + wave) * 16;
  const int bb = rowbase / Nn;
  const _Float16* kbase = kh  + (size_t)bb * Nn * E;
  const _Float16* vtb   = vtg + (size_t)bb * E * Nn;

  const int ve   = tid >> 1;  // V^T row this thread copies
  const int vseg = tid & 1;   // 32-key segment within the row

  // B = Q^T, loop-invariant (lane's query column, contiguous e run).
  v16h bq[2];
  bq[0] = load_b(qh + (size_t)(rowbase + n) * E + 0  + kg * 16);
  bq[1] = load_b(qh + (size_t)(rowbase + n) * E + 32 + kg * 16);

  v16h vones;
  #pragma unroll
  for (int i = 0; i < 16; ++i) vones[i] = (_Float16)1.0f;

  float mq = -3.0e38f, lq = 0.0f; // per-lane (per-query) online stats
  v8f ot[4];                       // O^T accumulators
  #pragma unroll
  for (int et = 0; et < 4; ++et)
    #pragma unroll
    for (int r = 0; r < 8; ++r) ot[et][r] = 0.0f;

  // Issue async fill helper: K natural (4x512B sweeps) + V^T rows.
  auto issue_fill = [&](int k0, int buf) {
    const _Float16* kg0 = kbase + (size_t)k0 * E;
    #pragma unroll
    for (int c = 0; c < 4; ++c) {
      int off = c * 2048 + tid * 16; // bytes within the 8 KB K block
      async_ld_b128((const char*)&kbuf[buf][0] + off, off, kg0);
    }
    const _Float16* vg0 = vtb + k0;
    #pragma unroll
    for (int j = 0; j < 4; ++j) {
      async_ld_b128(&vbuf[buf][ve * VTS + vseg * 32 + j * 8],
                    (ve * Nn + vseg * 32 + j * 8) * 2, vg0);
    }
  };

  constexpr int NIT = Nn / BK; // 64 iterations
  issue_fill(0, 0);

  for (int it = 0; it < NIT; ++it) {
    const int cur = it & 1;
    wait_async0();   // this wave's portion of buffer `cur` has landed
    __syncthreads(); // everyone's portion has landed; prior reads consumed
    if (it + 1 < NIT) issue_fill((it + 1) * BK, cur ^ 1);

    const _Float16* kb = &kbuf[cur][0];
    const _Float16* vt = &vbuf[cur][0];

    // S^T tiles: A = K rows (16 keys x 32 e), B = Q^T. 4 tiles x 2 k-steps.
    v8f st[4];
    #pragma unroll
    for (int ct = 0; ct < 4; ++ct) {
      v8f acc = {};
      #pragma unroll
      for (int ks = 0; ks < 2; ++ks) {
        v16h ak = load_a(kb + (ct * 16 + m) * E + ks * 32, half);
        acc = __builtin_amdgcn_wmma_f32_16x16x32_f16(
                  false, ak, false, bq[ks], (short)0, acc, false, false);
      }
      st[ct] = acc;
    }

    // Row (= lane) max over 32 in-lane values, then cross-half exchange.
    float m01[8], m23[8], mm[8];
    #pragma unroll
    for (int r = 0; r < 8; ++r) {
      m01[r] = fmaxf(st[0][r], st[1][r]);
      m23[r] = fmaxf(st[2][r], st[3][r]);
      mm[r]  = fmaxf(m01[r], m23[r]);
    }
    float a0 = fmaxf(fmaxf(mm[0], mm[1]), fmaxf(mm[2], mm[3]));
    float a1 = fmaxf(fmaxf(mm[4], mm[5]), fmaxf(mm[6], mm[7]));
    float sm = fmaxf(a0, a1);
    sm = fmaxf(sm, __shfl_xor(sm, 16, 32));

    float mnew = fmaxf(mq, sm);
    float corr = exp2f(mq - mnew); // base-2: scale folded into Q
    mq = mnew;

    // P^T = exp2(S^T - m): 8 consecutive keys per lane per tile -> b128 store.
    _Float16* pw = &ps[wave][0];
    #pragma unroll
    for (int ct = 0; ct < 4; ++ct) {
      v8h ph;
      #pragma unroll
      for (int r = 0; r < 8; ++r) ph[r] = (_Float16)exp2f(st[ct][r] - mq);
      *(v8h*)(pw + n * PWS + ct * 16 + 8 * half) = ph;
    }

    // P^T as reusable B operands (2 k-steps of 32 keys).
    v16h bp0 = load_b(pw + n * PWS + 0  + kg * 16);
    v16h bp1 = load_b(pw + n * PWS + 32 + kg * 16);

    // Column sums of P^T (= softmax row sums) via ones @ P^T.
    v8f rsum = {};
    rsum = __builtin_amdgcn_wmma_f32_16x16x32_f16(
               false, vones, false, bp0, (short)0, rsum, false, false);
    rsum = __builtin_amdgcn_wmma_f32_16x16x32_f16(
               false, vones, false, bp1, (short)0, rsum, false, false);
    lq = lq * corr + rsum[0];

    #pragma unroll
    for (int et = 0; et < 4; ++et)
      #pragma unroll
      for (int r = 0; r < 8; ++r) ot[et][r] *= corr;

    // O^T += V^T @ P^T : A = V^T rows (16 e x 32 keys), B = P^T.
    #pragma unroll
    for (int et = 0; et < 4; ++et) {
      v16h av0 = load_a(vt + (et * 16 + m) * VTS + 0, half);
      ot[et] = __builtin_amdgcn_wmma_f32_16x16x32_f16(
                   false, av0, false, bp0, (short)0, ot[et], false, false);
      v16h av1 = load_a(vt + (et * 16 + m) * VTS + 32, half);
      ot[et] = __builtin_amdgcn_wmma_f32_16x16x32_f16(
                   false, av1, false, bp1, (short)0, ot[et], false, false);
    }
  }

  // Normalize (one rcp per lane) and store: 8 consecutive e per tile.
  float inv = 1.0f / lq;
  #pragma unroll
  for (int et = 0; et < 4; ++et) {
    float4 lo = {ot[et][0] * inv, ot[et][1] * inv, ot[et][2] * inv, ot[et][3] * inv};
    float4 hi = {ot[et][4] * inv, ot[et][5] * inv, ot[et][6] * inv, ot[et][7] * inv};
    float* op = out + (size_t)(rowbase + n) * E + et * 16 + 8 * half;
    *(float4*)op       = lo;
    *(float4*)(op + 4) = hi;
  }
}

extern "C" void kernel_launch(void* const* d_in, const int* in_sizes, int n_in,
                              void* d_out, int out_size, void* d_ws, size_t ws_size,
                              hipStream_t stream) {
  (void)in_sizes; (void)n_in; (void)out_size; (void)ws_size;
  const float* x = (const float*)d_in[0];
  const float* w = (const float*)d_in[1];
  _Float16* qh  = (_Float16*)d_ws;        // Q  [row][e]  f16, 2 MB
  _Float16* kh  = qh + (size_t)ROWS * E;  // K  [row][e]  f16, 2 MB
  _Float16* vtg = kh + (size_t)ROWS * E;  // V^T[b][e][row] f16, 2 MB
  float* out = (float*)d_out;

  qkv_kernel<<<dim3(ROWS / 128), dim3(256), 0, stream>>>(x, w, qh, kh, vtg);
  attn_kernel<<<dim3(ROWS / 64), dim3(128), 0, stream>>>(qh, kh, vtg, out);
}
